// Point_contextual_hyper_prior_encoder_6047313953111
// MI455X (gfx1250) — matrix operand order repaired and checked by reading, not compiled, not run on hardware
//
#include <hip/hip_runtime.h>
#include <hip/hip_bf16.h>

// MI455X / gfx1250, wave32. D = A*B + C via v_wmma_f32_16x16x32_f16.
// Weight panels stream global->LDS with GLOBAL_LOAD_ASYNC_TO_LDS_B128 (ASYNCcnt).
typedef __attribute__((ext_vector_type(16))) _Float16 v16h;
typedef __attribute__((ext_vector_type(8)))  float    v8f;

#define LRELU_SLOPE 0.1f
#define BN_EPS 1e-5f

// ---- CDNA5 async global->LDS copy (ISA 15.18.3 op 98), GV addressing ----
__device__ __forceinline__ void async_g2l_b128(unsigned int lds_byte_addr,
                                               const void* gaddr) {
    asm volatile("global_load_async_to_lds_b128 %0, %1, off"
                 :: "v"(lds_byte_addr),
                    "v"((unsigned long long)(uintptr_t)gaddr)
                 : "memory");
}
template <int N>
__device__ __forceinline__ void wait_asynccnt() {
    asm volatile("s_wait_asynccnt %0" :: "n"(N) : "memory");
}

// ---------------------------------------------------------------------------
// Prep: W [KT][CIN][64] f32  ->  Wt [KT][64][CIN] f16  (transposed for B-frag)
// ---------------------------------------------------------------------------
__global__ __launch_bounds__(256) void w_prep_kernel(const float* __restrict__ W,
                                                     _Float16* __restrict__ Wt,
                                                     int KT, int CIN) {
    int total = KT * CIN * 64;
    for (int i = blockIdx.x * blockDim.x + threadIdx.x; i < total;
         i += gridDim.x * blockDim.x) {
        int k = i / (CIN * 64);
        int rem = i - k * CIN * 64;
        int c = rem >> 6;           // Cin index
        int d = rem & 63;           // Cout index
        Wt[((size_t)(k * 64 + d)) * CIN + c] = (_Float16)W[i];
    }
}

// x [N][C] f32 -> xh [N+1][C] f16, sentinel row N = zeros
__global__ __launch_bounds__(256) void x_prep_kernel(const float* __restrict__ x,
                                                     _Float16* __restrict__ xh,
                                                     int N, int C) {
    int total = (N + 1) * C;
    for (int i = blockIdx.x * blockDim.x + threadIdx.x; i < total;
         i += gridDim.x * blockDim.x) {
        int r = i / C;
        xh[i] = (r < N) ? (_Float16)x[i] : (_Float16)0.0f;
    }
}

__global__ void zero_row64_kernel(_Float16* __restrict__ p, int row) {
    p[(size_t)row * 64 + threadIdx.x] = (_Float16)0.0f;
}

// ---------------------------------------------------------------------------
// WMMA gather-GEMM: out[n,d] = sum_k sum_c featPad[nmap[n,k], c] * W[k,c,d] + b[d]
// One wave = 32 output rows x 64 output cols (2 row-tiles x 4 col-tiles of
// v8f accumulators). Per-tap weight panels are triple-buffered in LDS and
// streamed with async global->LDS copies: one barrier per tap, copy/compute
// overlap enforced by s_wait_asynccnt.
// ---------------------------------------------------------------------------
template <int CIN, int KT, bool LEAKY, bool TO_HALF>
__global__ __launch_bounds__(256) void sconv_wmma_kernel(
    const _Float16* __restrict__ feat,   // [NinPad+1][CIN] f16, row NinPad = 0
    const int*      __restrict__ nmap,   // [Nout][KT]
    const _Float16* __restrict__ wt,     // [KT][64][CIN] f16 (transposed)
    const float*    __restrict__ bias,   // [64]
    float*          __restrict__ outf,   // [Nout][64] f32   (if !TO_HALF)
    _Float16*       __restrict__ outh,   // [>=Nout][64] f16 (if TO_HALF)
    int Nout, int NinPad) {
    constexpr int PANEL_H = 64 * CIN;              // halves per weight panel
    constexpr int PANEL_B = PANEL_H * 2;           // bytes per weight panel
    constexpr int NV = PANEL_B / (16 * 256);       // 16B chunks per thread
    __shared__ __align__(16) _Float16 smw[3][PANEL_H];  // 36 KB / 24 KB

    const int tid  = threadIdx.x;
    const int wave = tid >> 5;
    const int lane = tid & 31;
    const int tile = blockIdx.x * 8 + wave;        // 32-row tile per wave
    const int m  = lane & 15;                      // row (A) / col (B,D) in tile
    const int hi = lane >> 4;                      // lane-half selector

    // Two row sub-tiles per wave (rows base..base+15 and base+16..base+31)
    const int r0 = tile * 32 + m;
    const int r1 = r0 + 16;
    const int rr0 = (r0 < Nout) ? r0 : (Nout - 1);
    const int rr1 = (r1 < Nout) ? r1 : (Nout - 1);

    v8f acc[2][4] = {};

    const char* wbase = (const char*)wt;
    // Per-buffer LDS byte addresses (low 32 bits of the flat LDS pointer)
    unsigned int lds_base[3];
#pragma unroll
    for (int b = 0; b < 3; ++b)
        lds_base[b] = (unsigned int)(uintptr_t)(&smw[b][0]);

    // ---- prologue: start async copy of tap-0 panel into buffer 0 ----
#pragma unroll
    for (int j = 0; j < NV; ++j) {
        int ci = tid + j * 256;
        async_g2l_b128(lds_base[0] + ci * 16, wbase + (size_t)ci * 16);
    }

    for (int k = 0; k < KT; ++k) {
        // Kick off the NEXT tap's panel into buffer (k+1)%3
        if (k + 1 < KT) {
            const char* src = wbase + (size_t)(k + 1) * PANEL_B;
            unsigned int dst = lds_base[(k + 1) % 3];
#pragma unroll
            for (int j = 0; j < NV; ++j) {
                int ci = tid + j * 256;
                async_g2l_b128(dst + ci * 16, src + (size_t)ci * 16);
            }
            wait_asynccnt<NV>();   // in-order: panel k done, k+1 may fly
        } else {
            wait_asynccnt<0>();
        }
        __syncthreads();           // panel k visible WGP-wide

        // Gather indices for this tap (sentinel = zero feature row)
        int idx0 = nmap[(size_t)rr0 * KT + k];
        int idx1 = nmap[(size_t)rr1 * KT + k];
        if (r0 >= Nout) idx0 = NinPad;
        if (r1 >= Nout) idx1 = NinPad;
        if ((unsigned)idx0 > (unsigned)NinPad) idx0 = NinPad;
        if ((unsigned)idx1 > (unsigned)NinPad) idx1 = NinPad;
        const _Float16* ap0 = feat + (size_t)idx0 * CIN + hi * 16;
        const _Float16* ap1 = feat + (size_t)idx1 * CIN + hi * 16;
        const _Float16* wp  = &smw[k % 3][0];

#pragma unroll
        for (int cb = 0; cb < CIN; cb += 32) {
            // A fragments: lane<16 -> K {cb..cb+15};
            //              lane>=16 -> K {cb+16..cb+31} (matches B contiguous order)
            v16h a0, a1;
            *((float4*)&a0)     = *(const float4*)(ap0 + cb);
            *(((float4*)&a0)+1) = *(const float4*)(ap0 + cb + 8);
            *((float4*)&a1)     = *(const float4*)(ap1 + cb);
            *(((float4*)&a1)+1) = *(const float4*)(ap1 + cb + 8);
#pragma unroll
            for (int n = 0; n < 4; ++n) {
                // B fragment from LDS: col d = n*16+m, 16 contiguous K-halves
                // at cb + 16*hi (contiguous k order, same as A; panel is [d][c])
                const _Float16* bp = wp + (n * 16 + m) * CIN + cb + hi * 16;
                v16h b;
                *((float4*)&b)     = *(const float4*)(bp);
                *(((float4*)&b)+1) = *(const float4*)(bp + 8);
                acc[0][n] = __builtin_amdgcn_wmma_f32_16x16x32_f16(
                    false, a0, false, b, (short)0, acc[0][n], false, false);
                acc[1][n] = __builtin_amdgcn_wmma_f32_16x16x32_f16(
                    false, a1, false, b, (short)0, acc[1][n], false, false);
            }
        }
        // No trailing barrier: triple buffering bounds wave skew to one tap.
    }

    // D layout: VGPR v -> row v + 8*hi, col = m (per 16x16 f32 C/D striping)
#pragma unroll
    for (int t = 0; t < 2; ++t) {
#pragma unroll
        for (int v = 0; v < 8; ++v) {
            const int orow = tile * 32 + t * 16 + v + hi * 8;
            if (orow < Nout) {
#pragma unroll
                for (int n = 0; n < 4; ++n) {
                    const int col = n * 16 + m;
                    float val = acc[t][n][v] + bias[col];
                    if (LEAKY) val = (val >= 0.0f) ? val : LRELU_SLOPE * val;
                    if (TO_HALF) outh[(size_t)orow * 64 + col] = (_Float16)val;
                    else         outf[(size_t)orow * 64 + col] = val;
                }
            }
        }
    }
}

// ---------------------------------------------------------------------------
// BatchNorm stats: one block per channel (deterministic, atomic-free).
// Produces folded coefficients: y = f*bnA[c] + bnB[c]  (includes gamma, beta,
// mean, rsqrt(var+eps) and the trailing scale).
// ---------------------------------------------------------------------------
__global__ __launch_bounds__(256) void bn_stats_kernel(
    const float* __restrict__ f2, int N2,
    const float* __restrict__ gamma, const float* __restrict__ beta,
    const float* __restrict__ scale,
    float* __restrict__ bnA, float* __restrict__ bnB) {
    __shared__ float ssum[256];
    __shared__ float ssq[256];
    const int c = blockIdx.x;  // 0..63
    float s = 0.0f, q = 0.0f;
    for (int r = threadIdx.x; r < N2; r += 256) {
        float v = f2[(size_t)r * 64 + c];
        s += v;
        q += v * v;
    }
    ssum[threadIdx.x] = s;
    ssq[threadIdx.x]  = q;
    __syncthreads();
    for (int st = 128; st > 0; st >>= 1) {
        if (threadIdx.x < st) {
            ssum[threadIdx.x] += ssum[threadIdx.x + st];
            ssq[threadIdx.x]  += ssq[threadIdx.x + st];
        }
        __syncthreads();
    }
    if (threadIdx.x == 0) {
        float inv = 1.0f / (float)N2;
        float mu  = ssum[0] * inv;
        float var = ssq[0] * inv - mu * mu;
        float rs  = rsqrtf(var + BN_EPS);
        float g   = gamma[c] * rs;
        float sc  = scale[0];
        bnA[c] = g * sc;
        bnB[c] = (beta[c] - mu * g) * sc;
    }
}

// f2 (f32) -> f2h (f16, normalized+scaled), sentinel row N2 = zeros
__global__ __launch_bounds__(256) void bn_apply_kernel(
    const float* __restrict__ f2, const float* __restrict__ bnA,
    const float* __restrict__ bnB, _Float16* __restrict__ f2h, int N2) {
    int total = (N2 + 1) * 64;
    for (int i = blockIdx.x * blockDim.x + threadIdx.x; i < total;
         i += gridDim.x * blockDim.x) {
        int r = i >> 6;
        int c = i & 63;
        f2h[i] = (r < N2) ? (_Float16)(f2[i] * bnA[c] + bnB[c]) : (_Float16)0.0f;
    }
}

// ---------------------------------------------------------------------------
extern "C" void kernel_launch(void* const* d_in, const int* in_sizes, int n_in,
                              void* d_out, int out_size, void* d_ws, size_t ws_size,
                              hipStream_t stream) {
    (void)n_in; (void)out_size; (void)ws_size;
    const float* x     = (const float*)d_in[0];
    const float* W1    = (const float*)d_in[1];
    const float* b1    = (const float*)d_in[2];
    const float* W2    = (const float*)d_in[3];
    const float* b2    = (const float*)d_in[4];
    const float* W3    = (const float*)d_in[5];
    const float* b3    = (const float*)d_in[6];
    const float* gamma = (const float*)d_in[7];
    const float* beta  = (const float*)d_in[8];
    const float* scale = (const float*)d_in[9];
    const int* nmap1   = (const int*)d_in[10];
    const int* nmap2   = (const int*)d_in[11];
    const int* nmap3   = (const int*)d_in[12];

    const int N  = in_sizes[0] / 96;   // stride-1 points
    const int N2 = in_sizes[11] / 8;   // stride-2 points

    // ---- workspace carve-up (256B aligned) ----
    char* base = (char*)d_ws;
    size_t off = 0;
    auto carve = [&](size_t bytes) -> char* {
        char* p = base + off;
        off = (off + bytes + 255) & ~(size_t)255;
        return p;
    };
    _Float16* wt1 = (_Float16*)carve((size_t)27 * 96 * 64 * 2);
    _Float16* wt2 = (_Float16*)carve((size_t)8  * 64 * 64 * 2);
    _Float16* wt3 = (_Float16*)carve((size_t)27 * 64 * 64 * 2);
    float*    bnA = (float*)carve(64 * 4);
    float*    bnB = (float*)carve(64 * 4);
    _Float16* xh  = (_Float16*)carve((size_t)(N + 1) * 96 * 2);
    _Float16* f1h = (_Float16*)carve((size_t)(N + 1) * 64 * 2);
    float*    f2  = (float*)carve((size_t)N2 * 64 * 4);
    _Float16* f2h = (_Float16*)carve((size_t)(N2 + 1) * 64 * 2);

    auto gs = [](int total) { int b = (total + 255) / 256; return b > 2048 ? 2048 : b; };

    // ---- prep: weight transpose->f16, x->f16 (+zero sentinel rows) ----
    w_prep_kernel<<<gs(27 * 96 * 64), 256, 0, stream>>>(W1, wt1, 27, 96);
    w_prep_kernel<<<gs(8 * 64 * 64),  256, 0, stream>>>(W2, wt2, 8, 64);
    w_prep_kernel<<<gs(27 * 64 * 64), 256, 0, stream>>>(W3, wt3, 27, 64);
    x_prep_kernel<<<gs((N + 1) * 96), 256, 0, stream>>>(x, xh, N, 96);
    zero_row64_kernel<<<1, 64, 0, stream>>>(f1h, N);

    // ---- conv1: k3 s1, 96->64, leaky, output f16 ----
    {
        int tiles = (N + 31) / 32;
        int blocks = (tiles + 7) / 8;
        sconv_wmma_kernel<96, 27, true, true><<<blocks, 256, 0, stream>>>(
            xh, nmap1, wt1, b1, (float*)nullptr, f1h, N, N);
    }
    // ---- conv2: k2 s2, 64->64, leaky, output f32 (for batch stats) ----
    {
        int tiles = (N2 + 31) / 32;
        int blocks = (tiles + 7) / 8;
        sconv_wmma_kernel<64, 8, true, false><<<blocks, 256, 0, stream>>>(
            f1h, nmap2, wt2, b2, f2, (_Float16*)nullptr, N2, N);
    }
    // ---- batchnorm (train-mode stats) + scale, folded to a*f+b ----
    bn_stats_kernel<<<64, 256, 0, stream>>>(f2, N2, gamma, beta, scale, bnA, bnB);
    bn_apply_kernel<<<gs((N2 + 1) * 64), 256, 0, stream>>>(f2, bnA, bnB, f2h, N2);

    // ---- conv3: k3 s1, 64->64, no activation, output f32 -> d_out ----
    {
        int tiles = (N2 + 31) / 32;
        int blocks = (tiles + 7) / 8;
        sconv_wmma_kernel<64, 27, false, false><<<blocks, 256, 0, stream>>>(
            f2h, nmap3, wt3, b3, (float*)d_out, (_Float16*)nullptr, N2, N2);
    }
}
